// DMPNNEncoder_54649163874391
// MI455X (gfx1250) — matrix operand order repaired and checked by reading, not compiled, not run on hardware
//
#include <hip/hip_runtime.h>
#include <stdint.h>

#define N_ATOMS   100000
#define N_BONDS   200000
#define MAX_NB    6
#define HIDDEN    300
#define ATOM_FD   133
#define BOND_FD   147
#define N_MOLS    5000
#define APM       20
#define CAT_FD    433          // ATOM_FD + HIDDEN

#define HP        320          // padded hidden (bf16 cols), 10 k-tiles of 32
#define HPP       160          // u32 pairs per padded hidden row
#define BFP       160          // padded bond-feature K (5 k-tiles)
#define BFPP      80           // u32 pairs per padded bond-feature row
#define CATP      448          // padded concat K (14 k-tiles)
#define CATPP     224          // u32 pairs per padded concat row
#define NT        20           // 320/16 n-tiles
#define KT_WI     5
#define KT_WH     10
#define KT_WO     14

typedef __attribute__((ext_vector_type(16))) __bf16 v16bf;
typedef __attribute__((ext_vector_type(8)))  float  v8f;

union BFrag { v16bf v; uint32_t u[8]; uint4 q[2]; };

__device__ __forceinline__ float bf2f(uint32_t hbits) {
    return __uint_as_float(hbits << 16);
}
__device__ __forceinline__ uint16_t f2bf(float f) {
    uint32_t u = __float_as_uint(f);
    u += 0x7FFFu + ((u >> 16) & 1u);           // round-to-nearest-even
    return (uint16_t)(u >> 16);
}
__device__ __forceinline__ uint32_t pk2(float lo, float hi) {
    return (uint32_t)f2bf(lo) | ((uint32_t)f2bf(hi) << 16);
}
__device__ __forceinline__ uint32_t subpk(uint32_t a, uint32_t b) {
    return pk2(bf2f(a & 0xFFFFu) - bf2f(b & 0xFFFFu),
               bf2f(a >> 16)     - bf2f(b >> 16));
}

// ---------------------------------------------------------------------------
// Pack fp32 weight [Ksrc x 300] into WMMA B-fragment order, bf16, zero-padded.
// Layout: (((kt*NT + nt)*32 + lane)*16 + e); lanes 0-15 K=k0..k0+7,k0+16..k0+23,
// lanes 16-31 K=k0+8..k0+15,k0+24..k0+31, two bf16 per dword (even K in [15:0]).
// ---------------------------------------------------------------------------
__global__ void pack_w(const float* __restrict__ src, int Ksrc, int ktiles,
                       uint16_t* __restrict__ dst) {
    int t = blockIdx.x * blockDim.x + threadIdx.x;
    int total = ktiles * NT * 512;
    if (t >= total) return;
    int e    = t & 15;
    int lane = (t >> 4) & 31;
    int nt   = (t >> 9) % NT;
    int kt   = t / (NT * 512);
    int k    = kt * 32 + (e >> 3) * 16 + (lane >> 4) * 8 + (e & 7);
    int col  = nt * 16 + (lane & 15);
    float v  = (k < Ksrc && col < HIDDEN) ? src[(size_t)k * HIDDEN + col] : 0.f;
    dst[t] = f2bf(v);
}

// f_bonds fp32 [N_BONDS x 147]  ->  bf16 padded [N_BONDS x 160]
__global__ void prep_bonds(const float* __restrict__ fb, uint32_t* __restrict__ dst) {
    int t = blockIdx.x * blockDim.x + threadIdx.x;
    if (t >= N_BONDS * BFPP) return;
    int b = t / BFPP, p = t - b * BFPP;
    int k = 2 * p;
    const float* row = fb + (size_t)b * BOND_FD;
    float v0 = (k     < BOND_FD) ? row[k]     : 0.f;
    float v1 = (k + 1 < BOND_FD) ? row[k + 1] : 0.f;
    dst[t] = pk2(v0, v1);
}

// [f_atoms fp32 | a_message bf16]  ->  bf16 padded [N_ATOMS x 448]
__global__ void prep_cat(const float* __restrict__ fa,
                         const uint16_t* __restrict__ amsg16,
                         uint32_t* __restrict__ dst) {
    int t = blockIdx.x * blockDim.x + threadIdx.x;
    if (t >= N_ATOMS * CATPP) return;
    int a = t / CATPP, p = t - a * CATPP;
    int k = 2 * p;
    const float*    arow = fa     + (size_t)a * ATOM_FD;
    const uint16_t* mrow = amsg16 + (size_t)a * HP;
    float v0 = (k < ATOM_FD) ? arow[k]
             : (k < CAT_FD ? bf2f(mrow[k - ATOM_FD]) : 0.f);
    int k1 = k + 1;
    float v1 = (k1 < ATOM_FD) ? arow[k1]
             : (k1 < CAT_FD ? bf2f(mrow[k1 - ATOM_FD]) : 0.f);
    dst[t] = pk2(v0, v1);
}

// ---------------------------------------------------------------------------
// Wave-level WMMA mainloop: 1 wave32 per block, 16(M) x 64(N) output tile,
// 4 v8f accumulators, branch-free b128 A/B loads.
// ---------------------------------------------------------------------------
__device__ __forceinline__ void load_bfrag(BFrag& b, const uint16_t* pW,
                                           int kt, int nt, int lane) {
    const uint4* bq = (const uint4*)pW;
    int bi = ((kt * NT + nt) * 32 + lane) * 2;
    b.q[0] = bq[bi];
    b.q[1] = bq[bi + 1];
}

__device__ __forceinline__ void wmma_loop(const uint4* __restrict__ arow,
                                          const uint16_t* __restrict__ pW,
                                          int ktiles, int nt0, int lane, int half,
                                          v8f c[4]) {
    for (int kt = 0; kt < ktiles; ++kt) {
        BFrag a;
        a.q[0] = arow[kt * 4 + half];
        a.q[1] = arow[kt * 4 + 2 + half];
        #pragma unroll
        for (int j = 0; j < 4; ++j) {
            BFrag b; load_bfrag(b, pW, kt, nt0 + j, lane);
            c[j] = __builtin_amdgcn_wmma_f32_16x16x32_bf16(
                       false, a.v, false, b.v, (short)0, c[j], false, false);
        }
    }
}

// inp = f_bonds @ W_i ; message = relu(inp); both bf16 padded to 320.
__global__ void gemm_in(const uint32_t* __restrict__ fbp,
                        const uint16_t* __restrict__ pW,
                        uint16_t* __restrict__ inp,
                        uint16_t* __restrict__ msg) {
    int lane = threadIdx.x, half = lane >> 4, lrow = lane & 15;
    int m0 = blockIdx.x * 16, nt0 = blockIdx.y * 4;
    const uint4* arow = (const uint4*)fbp + (size_t)(m0 + lrow) * (BFPP / 4);
    v8f c[4] = {};
    wmma_loop(arow, pW, KT_WI, nt0, lane, half, c);
    #pragma unroll
    for (int j = 0; j < 4; ++j) {
        #pragma unroll
        for (int v = 0; v < 8; ++v) {
            int r = m0 + half * 8 + v;
            int col = (nt0 + j) * 16 + lrow;
            size_t idx = (size_t)r * HP + col;
            float x = c[j][v];
            inp[idx] = f2bf(x);
            msg[idx] = f2bf(fmaxf(x, 0.f));
        }
    }
}

// message_new = relu(inp + (a_msg[b2a] - msg_old[b2revb]) @ W_h)
// Bond-level gather + subtract fused into the A-fragment loader (branch-free).
__global__ void gemm_h(const uint16_t* __restrict__ inp,
                       const uint32_t* __restrict__ amsg,
                       const uint32_t* __restrict__ msg_old,
                       const int* __restrict__ b2a,
                       const int* __restrict__ b2revb,
                       const uint16_t* __restrict__ pW,
                       uint16_t* __restrict__ msg_new) {
    int lane = threadIdx.x, half = lane >> 4, lrow = lane & 15;
    int m0 = blockIdx.x * 16, nt0 = blockIdx.y * 4;
    int row = m0 + lrow;
    const uint4* aq = (const uint4*)(amsg    + (size_t)b2a[row]    * HPP);
    const uint4* mq = (const uint4*)(msg_old + (size_t)b2revb[row] * HPP);
    v8f c[4] = {};
    for (int kt = 0; kt < KT_WH; ++kt) {
        BFrag a;
        uint4 x0 = aq[kt * 4 + half],     y0 = mq[kt * 4 + half];
        uint4 x1 = aq[kt * 4 + 2 + half], y1 = mq[kt * 4 + 2 + half];
        a.u[0] = subpk(x0.x, y0.x); a.u[1] = subpk(x0.y, y0.y);
        a.u[2] = subpk(x0.z, y0.z); a.u[3] = subpk(x0.w, y0.w);
        a.u[4] = subpk(x1.x, y1.x); a.u[5] = subpk(x1.y, y1.y);
        a.u[6] = subpk(x1.z, y1.z); a.u[7] = subpk(x1.w, y1.w);
        #pragma unroll
        for (int j = 0; j < 4; ++j) {
            BFrag b; load_bfrag(b, pW, kt, nt0 + j, lane);
            c[j] = __builtin_amdgcn_wmma_f32_16x16x32_bf16(
                       false, a.v, false, b.v, (short)0, c[j], false, false);
        }
    }
    #pragma unroll
    for (int j = 0; j < 4; ++j) {
        #pragma unroll
        for (int v = 0; v < 8; ++v) {
            int r = m0 + half * 8 + v;
            int col = (nt0 + j) * 16 + lrow;
            size_t idx = (size_t)r * HP + col;
            float x = bf2f(inp[idx]) + c[j][v];
            msg_new[idx] = f2bf(fmaxf(x, 0.f));
        }
    }
}

// atom_hiddens = relu(catA @ W_o + b_o)   (fp32 out, stride 300)
__global__ void gemm_o(const uint32_t* __restrict__ catA,
                       const uint16_t* __restrict__ pW,
                       const float* __restrict__ b_o,
                       float* __restrict__ hid) {
    int lane = threadIdx.x, half = lane >> 4, lrow = lane & 15;
    int m0 = blockIdx.x * 16, nt0 = blockIdx.y * 4;
    const uint4* arow = (const uint4*)catA + (size_t)(m0 + lrow) * (CATPP / 4);
    v8f c[4] = {};
    wmma_loop(arow, pW, KT_WO, nt0, lane, half, c);
    #pragma unroll
    for (int j = 0; j < 4; ++j) {
        int col = (nt0 + j) * 16 + lrow;
        if (col < HIDDEN) {
            float bias = b_o[col];
            #pragma unroll
            for (int v = 0; v < 8; ++v) {
                int r = m0 + half * 8 + v;
                hid[(size_t)r * HIDDEN + col] = fmaxf(c[j][v] + bias, 0.f);
            }
        }
    }
}

// a_message[a] = sum_{j<6} message[a2b[a][j]]   (bf16 pairs, fp32 accumulate)
__global__ void gather_sum(const int* __restrict__ a2b,
                           const uint32_t* __restrict__ msg,
                           uint32_t* __restrict__ amsg) {
    int t = blockIdx.x * blockDim.x + threadIdx.x;
    if (t >= N_ATOMS * HPP) return;
    int a = t / HPP, p = t - a * HPP;
    float s0 = 0.f, s1 = 0.f;
    #pragma unroll
    for (int j = 0; j < MAX_NB; ++j) {
        int b = a2b[a * MAX_NB + j];
        uint32_t u = msg[(size_t)b * HPP + p];
        s0 += bf2f(u & 0xFFFFu);
        s1 += bf2f(u >> 16);
    }
    amsg[t] = pk2(s0, s1);
}

// mean pool over contiguous 20-atom molecules
__global__ void pool(const float* __restrict__ hid, float* __restrict__ out) {
    int t = blockIdx.x * blockDim.x + threadIdx.x;
    if (t >= N_MOLS * HIDDEN) return;
    int m = t / HIDDEN, h = t - m * HIDDEN;
    float s = 0.f;
    #pragma unroll
    for (int a = 0; a < APM; ++a)
        s += hid[((size_t)m * APM + a) * HIDDEN + h];
    out[t] = s * (1.f / APM);
}

extern "C" void kernel_launch(void* const* d_in, const int* in_sizes, int n_in,
                              void* d_out, int out_size, void* d_ws, size_t ws_size,
                              hipStream_t stream) {
    const float* f_atoms = (const float*)d_in[0];
    const float* f_bonds = (const float*)d_in[1];
    const int*   a2b     = (const int*)d_in[2];
    const int*   b2a     = (const int*)d_in[3];
    const int*   b2revb  = (const int*)d_in[4];
    /* d_in[5] mol_ids unused: scopes are contiguous equal-size */
    const float* W_i     = (const float*)d_in[6];
    const float* W_h     = (const float*)d_in[7];
    const float* W_o     = (const float*)d_in[8];
    const float* b_o     = (const float*)d_in[9];
    float* out = (float*)d_out;

    char* ws = (char*)d_ws;
    const size_t SZ_MSG  = (size_t)N_BONDS * HP * 2;       // 128 MB
    const size_t SZ_AMSG = (size_t)N_ATOMS * HP * 2;       //  64 MB
    const size_t SZ_HID  = (size_t)N_ATOMS * HIDDEN * 4;   // 120 MB
    const size_t SZ_AUX  = (size_t)N_ATOMS * CATP * 2;     // 89.6 MB (>= bonds prep 64 MB)
    uint16_t* inp  = (uint16_t*)(ws);
    uint16_t* msgA = (uint16_t*)(ws + SZ_MSG);
    uint16_t* msgB = (uint16_t*)(ws + 2 * SZ_MSG);
    uint16_t* amsg = (uint16_t*)(ws + 3 * SZ_MSG);
    float*    hid  = (float*)   (ws + 3 * SZ_MSG + SZ_AMSG);
    uint32_t* aux  = (uint32_t*)(ws + 3 * SZ_MSG + SZ_AMSG + SZ_HID); // fbp / catA alias
    uint16_t* pWi  = (uint16_t*)(ws + 3 * SZ_MSG + SZ_AMSG + SZ_HID + SZ_AUX);
    uint16_t* pWh  = pWi + (size_t)KT_WI * NT * 512;
    uint16_t* pWo  = pWh + (size_t)KT_WH * NT * 512;

    // weight packing + bond-feature bf16 prep
    {
        int t1 = KT_WI * NT * 512, t2 = KT_WH * NT * 512, t3 = KT_WO * NT * 512;
        pack_w<<<(t1 + 255) / 256, 256, 0, stream>>>(W_i, BOND_FD, KT_WI, pWi);
        pack_w<<<(t2 + 255) / 256, 256, 0, stream>>>(W_h, HIDDEN,  KT_WH, pWh);
        pack_w<<<(t3 + 255) / 256, 256, 0, stream>>>(W_o, CAT_FD,  KT_WO, pWo);
        prep_bonds<<<(N_BONDS * BFPP + 255) / 256, 256, 0, stream>>>(f_bonds, aux);
    }

    dim3 gB(N_BONDS / 16, 5);   // 12500 x 5 wave-tiles (16M x 64N)
    dim3 gA(N_ATOMS / 16, 5);   //  6250 x 5
    int gsBlocks = (N_ATOMS * HPP + 255) / 256;

    gemm_in<<<gB, 32, 0, stream>>>(aux, pWi, inp, msgA);

    // depth-1 = 2 message-passing iterations (ping-pong msgA/msgB)
    gather_sum<<<gsBlocks, 256, 0, stream>>>(a2b, (const uint32_t*)msgA, (uint32_t*)amsg);
    gemm_h<<<gB, 32, 0, stream>>>(inp, (const uint32_t*)amsg, (const uint32_t*)msgA,
                                  b2a, b2revb, pWh, msgB);
    gather_sum<<<gsBlocks, 256, 0, stream>>>(a2b, (const uint32_t*)msgB, (uint32_t*)amsg);
    gemm_h<<<gB, 32, 0, stream>>>(inp, (const uint32_t*)amsg, (const uint32_t*)msgB,
                                  b2a, b2revb, pWh, msgA);

    // readout
    gather_sum<<<gsBlocks, 256, 0, stream>>>(a2b, (const uint32_t*)msgA, (uint32_t*)amsg);
    prep_cat<<<(N_ATOMS * CATPP + 255) / 256, 256, 0, stream>>>(f_atoms, amsg, aux);
    gemm_o<<<gA, 32, 0, stream>>>(aux, pWo, b_o, hid);
    pool<<<(N_MOLS * HIDDEN + 255) / 256, 256, 0, stream>>>(hid, out);
}
